// MutiHeadAttention_8761733284323
// MI455X (gfx1250) — matrix operand order repaired and checked by reading, not compiled, not run on hardware
//
#include <hip/hip_runtime.h>
#include <hip/hip_bf16.h>
#include <math.h>

// ---------------- problem constants ----------------
#define B_    4
#define SEQ   2048
#define DM    512
#define H_    4
#define DK    64
#define DV    128
#define HDK   (H_ * DK)   // 256
#define HDV   (H_ * DV)   // 512
#define MTOT  (B_ * SEQ)  // 8192

// ---------------- WMMA fragment types ----------------
typedef __attribute__((ext_vector_type(16))) __bf16 v16bf;
typedef __attribute__((ext_vector_type(8)))  __bf16 v8bf;
typedef __attribute__((ext_vector_type(8)))  float  v8f;

union Frag {
    v16bf v;
    v8bf  h[2];
};

__device__ __forceinline__ v8f wmma_bf16(const Frag& a, const Frag& b, v8f c) {
    return __builtin_amdgcn_wmma_f32_16x16x32_bf16(
        /*neg_a=*/false, a.v, /*neg_b=*/false, b.v,
        /*c_mod=*/(short)0, c, /*reuse_a=*/false, /*reuse_b=*/false);
}

// ---------------- conversion kernels ----------------
__global__ void cvt_f32_bf16(const float* __restrict__ in, __bf16* __restrict__ out, int n) {
    int i = blockIdx.x * blockDim.x + threadIdx.x;
    if (i < n) out[i] = (__bf16)in[i];
}

// in: [K, Nw] row-major f32  ->  out: [Nw, K] row-major bf16 (i.e. transposed)
__global__ void cvt_transpose_bf16(const float* __restrict__ in, __bf16* __restrict__ out,
                                   int K, int Nw) {
    int i = blockIdx.x * blockDim.x + threadIdx.x;
    if (i >= K * Nw) return;
    int k = i / Nw;
    int n = i - k * Nw;
    out[(size_t)n * K + k] = (__bf16)in[i];
}

// ---------------- generic WMMA GEMM ----------------
// C[M,Nw] = A[M,K] (bf16 row-major) * W (bf16, stored TRANSPOSED as Wt[Nw,K] row-major)
// OUTMODE: 0 = bf16 row-major, 1 = bf16 transposed ([Nw, M]), 2 = f32 row-major
// Block = 128 threads (4 waves). Wave computes a 16(M) x 64(N) tile.
template <int OUTMODE>
__global__ void gemm16x64(const __bf16* __restrict__ A, const __bf16* __restrict__ Wt,
                          void* __restrict__ Cout, int M, int Nw, int K) {
    const int lane = threadIdx.x & 31;
    const int wave = threadIdx.x >> 5;
    const int m0   = blockIdx.x * 16;
    const int n0   = blockIdx.y * 256 + wave * 64;
    if (n0 >= Nw) return;

    const int mrow = lane & 15;   // A row / B column within tile
    const int half = lane >> 4;   // K-half selector

    v8f acc[4] = {v8f{}, v8f{}, v8f{}, v8f{}};

    for (int k0 = 0; k0 < K; k0 += 32) {
        Frag a;
        const __bf16* ap = A + (size_t)(m0 + mrow) * K + k0 + half * 8;
        a.h[0] = *(const v8bf*)ap;         // K = half*8 + 0..7
        a.h[1] = *(const v8bf*)(ap + 16);  // K = half*8 + 16..23
#pragma unroll
        for (int s = 0; s < 4; ++s) {
            Frag bf;
            const __bf16* bp = Wt + (size_t)(n0 + s * 16 + mrow) * K + k0 + half * 8;
            bf.h[0] = *(const v8bf*)bp;        // K = half*8 + 0..7
            bf.h[1] = *(const v8bf*)(bp + 16); // K = half*8 + 16..23
            acc[s] = wmma_bf16(a, bf, acc[s]);
        }
    }

#pragma unroll
    for (int s = 0; s < 4; ++s) {
#pragma unroll
        for (int r = 0; r < 8; ++r) {
            int row = m0 + r + 8 * half;
            int col = n0 + s * 16 + mrow;
            float val = acc[s][r];
            if (OUTMODE == 0) {
                ((__bf16*)Cout)[(size_t)row * Nw + col] = (__bf16)val;
            } else if (OUTMODE == 1) {
                ((__bf16*)Cout)[(size_t)col * M + row] = (__bf16)val;
            } else {
                ((float*)Cout)[(size_t)row * Nw + col] = val;
            }
        }
    }
}

// ---------------- batch-softmax attention ----------------
// Q,K: [MTOT, HDK] bf16 row-major (row = b*SEQ+n, col = h*DK+d)
// Vt:  [HDV, MTOT] bf16 row-major (transposed V; row = h*DV+dv, col = b*SEQ+n)
// Ctx: [MTOT, HDV] bf16 row-major
// Block = 128 threads (4 waves); block handles (h, 16-query tile); wave owns 32 dv lanes.
// softmax is over the batch axis (4 values per (h,q,k)) -> purely lane-local.
//
// Q tile lives in LDS. The A-fragment loads from it are indexed with an opaque
// zero re-defined *inside* the k-loop by an empty `asm volatile` (zero cost):
// this blocks LICM from hoisting 64 VGPRs of loop-invariant fragments out of
// the loop (which caused scratch spills), while keeping them ordinary
// addrspace(3) ds_load_b128s with compiler-managed DScnt waits.
// amdgpu_waves_per_eu(1) lets the allocator use the full VGPR file and kill
// the last per-iteration spill slot; the grid still provides multiple blocks
// per WGP and the load clauses give ample latency tolerance.
__global__ void __attribute__((amdgpu_waves_per_eu(1)))
attn_bsm(const __bf16* __restrict__ Q, const __bf16* __restrict__ Km,
         const __bf16* __restrict__ Vt, __bf16* __restrict__ Ctx) {
    const int lane = threadIdx.x & 31;
    const int wave = threadIdx.x >> 5;   // dv chunk of 32
    const int h    = blockIdx.y;
    const int m0   = blockIdx.x * 16;    // query tile base (within each batch's SEQ)
    const int mrow = lane & 15;
    const int half = lane >> 4;

    __shared__ __align__(16) __bf16 Qs[B_][16][DK];        // 8 KB: Q tile, all batches
    __shared__ __align__(16) __bf16 Pbuf[4][B_][16 * 32];  // 16 KB: [wave][b][16q x 32k]

    // ---- stage Q tile into LDS (64 rows x 128 B; 128 threads take half-rows) ----
    {
        int t    = threadIdx.x;
        int row  = t >> 1;       // 0..63 = b*16 + m
        int part = t & 1;        // which 64-byte half of the row
        int b = row >> 4, m = row & 15;
        const uint4* src = (const uint4*)(Q + (size_t)(b * SEQ + m0 + m) * HDK + h * DK) + part * 4;
        uint4* dst = (uint4*)(&Qs[b][m][0]) + part * 4;
        dst[0] = src[0];
        dst[1] = src[1];
        dst[2] = src[2];
        dst[3] = src[3];
    }
    __syncthreads();

    v8f o[B_][2];
#pragma unroll
    for (int b = 0; b < B_; ++b) { o[b][0] = v8f{}; o[b][1] = v8f{}; }

    unsigned qzero = 0;  // opaque 0, refreshed every k-iteration

    for (int kt = 0; kt < SEQ; kt += 32) {
        // Re-define qzero inside the loop; compiler must assume it changes.
        asm volatile("" : "+v"(qzero));

        // ---- scores + batch softmax, one 16-wide kseq subtile at a time ----
#pragma unroll
        for (int sub = 0; sub < 2; ++sub) {
            v8f sv[B_];
#pragma unroll
            for (int b = 0; b < B_; ++b) {
                v8f s = v8f{};
#pragma unroll
                for (int c = 0; c < 2; ++c) {   // dk chunks of 32
                    Frag qf;  // A-fragment from LDS Q tile (non-hoistable via qzero)
                    const __bf16* qp = &Qs[b][mrow][c * 32 + half * 8 + qzero];
                    qf.h[0] = *(const v8bf*)qp;         // K = half*8 + 0..7
                    qf.h[1] = *(const v8bf*)(qp + 16);  // K = half*8 + 16..23
                    Frag kf;  // B = K^T: column = kseq row of K -> contiguous dk
                    const __bf16* kp = Km + (size_t)(b * SEQ + kt + sub * 16 + mrow) * HDK
                                       + h * DK + c * 32 + half * 8;
                    kf.h[0] = *(const v8bf*)kp;
                    kf.h[1] = *(const v8bf*)(kp + 16);
                    s = wmma_bf16(qf, kf, s);
                }
                sv[b] = s;
            }
            // softmax over the batch axis (4 values, lane-local)
#pragma unroll
            for (int r = 0; r < 8; ++r) {
                float x0 = sv[0][r] * 0.125f;   // 1/sqrt(64)
                float x1 = sv[1][r] * 0.125f;
                float x2 = sv[2][r] * 0.125f;
                float x3 = sv[3][r] * 0.125f;
                float mx = fmaxf(fmaxf(x0, x1), fmaxf(x2, x3));
                float e0 = __expf(x0 - mx);
                float e1 = __expf(x1 - mx);
                float e2 = __expf(x2 - mx);
                float e3 = __expf(x3 - mx);
                float inv = 1.0f / (e0 + e1 + e2 + e3);
                int row = r + 8 * half;         // 0..15
                int col = sub * 16 + mrow;      // 0..31
                Pbuf[wave][0][row * 32 + col] = (__bf16)(e0 * inv);
                Pbuf[wave][1][row * 32 + col] = (__bf16)(e1 * inv);
                Pbuf[wave][2][row * 32 + col] = (__bf16)(e2 * inv);
                Pbuf[wave][3][row * 32 + col] = (__bf16)(e3 * inv);
            }
        }

        // ---- O_b += P_b[16x32] @ V_b[32k x 32dv-chunk] ----
        // (Pbuf is wave-private; DS in-order within wave handles store->load)
#pragma unroll
        for (int b = 0; b < B_; ++b) {
            Frag pf;  // A-fragment layout from row-major P in LDS
            const __bf16* pp = &Pbuf[wave][b][mrow * 32 + half * 8];
            pf.h[0] = *(const v8bf*)pp;
            pf.h[1] = *(const v8bf*)(pp + 16);
#pragma unroll
            for (int nt = 0; nt < 2; ++nt) {
                Frag vf;  // B column = dv lane -> row of Vt -> contiguous over k
                const __bf16* vp = Vt + (size_t)(h * DV + wave * 32 + nt * 16 + mrow) * MTOT
                                   + b * SEQ + kt + half * 8;
                vf.h[0] = *(const v8bf*)vp;
                vf.h[1] = *(const v8bf*)(vp + 16);
                o[b][nt] = wmma_bf16(pf, vf, o[b][nt]);
            }
        }
    }

    // ---- epilogue: Ctx[b*SEQ+q][h*DV + dv] ----
#pragma unroll
    for (int b = 0; b < B_; ++b)
#pragma unroll
        for (int nt = 0; nt < 2; ++nt)
#pragma unroll
            for (int r = 0; r < 8; ++r) {
                int row = b * SEQ + m0 + r + 8 * half;
                int col = h * DV + wave * 32 + nt * 16 + mrow;
                Ctx[(size_t)row * HDV + col] = (__bf16)o[b][nt][r];
            }
}

// ---------------- host-side orchestration ----------------
extern "C" void kernel_launch(void* const* d_in, const int* in_sizes, int n_in,
                              void* d_out, int out_size, void* d_ws, size_t ws_size,
                              hipStream_t stream) {
    const float* x  = (const float*)d_in[0];  // [B, N, DM]
    const float* Wq = (const float*)d_in[1];  // [DM, HDK]
    const float* Wk = (const float*)d_in[2];  // [DM, HDK]
    const float* Wv = (const float*)d_in[3];  // [DM, HDV]
    const float* Wo = (const float*)d_in[4];  // [DM, DM]
    float* out = (float*)d_out;               // [B, N, DM]

    // workspace carve-up (bytes)
    char* ws = (char*)d_ws;
    __bf16* Xb  = (__bf16*)(ws);                          // 8192*512   = 8 MB
    __bf16* Wqt = (__bf16*)(ws + (8u << 20));             // 256*512    = 256 KB
    __bf16* Wkt = (__bf16*)(ws + (8u << 20) + (256u << 10));
    __bf16* Wvt = (__bf16*)(ws + (8u << 20) + (512u << 10));   // 512*512 = 512 KB
    __bf16* Wot = (__bf16*)(ws + (9u << 20));
    __bf16* Qb  = (__bf16*)(ws + (9u << 20) + (512u << 10));   // 8192*256 = 4 MB
    __bf16* Kb  = (__bf16*)(ws + (13u << 20) + (512u << 10));  // 4 MB
    __bf16* Vt  = (__bf16*)(ws + (17u << 20) + (512u << 10));  // 512*8192 = 8 MB (transposed)
    __bf16* Ctx = (__bf16*)(ws + (25u << 20) + (512u << 10));  // 8192*512 = 8 MB

    // 1) precision conversion (+ weight transposes for contiguous B-fragments)
    {
        int n = MTOT * DM;
        cvt_f32_bf16<<<(n + 255) / 256, 256, 0, stream>>>(x, Xb, n);
        cvt_transpose_bf16<<<(DM * HDK + 255) / 256, 256, 0, stream>>>(Wq, Wqt, DM, HDK);
        cvt_transpose_bf16<<<(DM * HDK + 255) / 256, 256, 0, stream>>>(Wk, Wkt, DM, HDK);
        cvt_transpose_bf16<<<(DM * HDV + 255) / 256, 256, 0, stream>>>(Wv, Wvt, DM, HDV);
        cvt_transpose_bf16<<<(DM * DM  + 255) / 256, 256, 0, stream>>>(Wo, Wot, DM, DM);
    }

    // 2) projections: Q, K row-major bf16; V transposed bf16
    gemm16x64<0><<<dim3(MTOT / 16, 1), 128, 0, stream>>>(Xb, Wqt, Qb, MTOT, HDK, DM);
    gemm16x64<0><<<dim3(MTOT / 16, 1), 128, 0, stream>>>(Xb, Wkt, Kb, MTOT, HDK, DM);
    gemm16x64<1><<<dim3(MTOT / 16, 2), 128, 0, stream>>>(Xb, Wvt, Vt, MTOT, HDV, DM);

    // 3) batch-axis-softmax attention -> Ctx
    attn_bsm<<<dim3(SEQ / 16, H_), 128, 0, stream>>>(Qb, Kb, Vt, Ctx);

    // 4) output projection (f32 out)
    gemm16x64<2><<<dim3(MTOT / 16, 2), 128, 0, stream>>>(Ctx, Wot, out, MTOT, DM, DM);
}
